// BitPackedGPT2_77558519431326
// MI455X (gfx1250) — compile-verified, matrix operand order checked by reading, not run on hardware
//
#include <hip/hip_runtime.h>
#include <hip/hip_bf16.h>

typedef _Float16 v16h __attribute__((ext_vector_type(16)));
typedef _Float16 v8h  __attribute__((ext_vector_type(8)));
typedef float    v8f  __attribute__((ext_vector_type(8)));
typedef int      v4i  __attribute__((ext_vector_type(4)));

#define S_  1024
#define D_  768
#define H_  12
#define HD_ 64
#define L_  12
#define F_  3072
#define V_  50257

#if defined(__has_builtin)
#if __has_builtin(__builtin_amdgcn_global_load_async_to_lds_b128)
#define ASYNC_LDS 1
#endif
#endif
#ifndef ASYNC_LDS
#define ASYNC_LDS 0
#endif

#if ASYNC_LDS
typedef __attribute__((address_space(1))) v4i* gv4p;  // global int4*
typedef __attribute__((address_space(3))) v4i* lv4p;  // LDS int4*
static __device__ __forceinline__ void wait_async_le2() {
#if __has_builtin(__builtin_amdgcn_s_wait_asynccnt)
  __builtin_amdgcn_s_wait_asynccnt(2);
#else
  asm volatile("s_wait_asynccnt 0x2" ::: "memory");
#endif
}
static __device__ __forceinline__ void wait_async_le0() {
#if __has_builtin(__builtin_amdgcn_s_wait_asynccnt)
  __builtin_amdgcn_s_wait_asynccnt(0);
#else
  asm volatile("s_wait_asynccnt 0x0" ::: "memory");
#endif
}
#endif

static __device__ __forceinline__ v8f wmma_f16(v16h a, v16h b, v8f c) {
  return __builtin_amdgcn_wmma_f32_16x16x32_f16(false, a, false, b, (short)0, c,
                                                false, false);
}

static __device__ __forceinline__ float gelu_f(float v) {
  // 0.5v(1+tanh(u)) == v*sigmoid(2u), u = 0.79788456(v + 0.044715 v^3)
  return v / (1.f + __expf(-1.5957691216057308f * (v + 0.044715f * v * v * v)));
}

struct CFalse { static constexpr bool value = false; };
struct CTrue  { static constexpr bool value = true;  };

// ---------------------------------------------------------------------------
// Embedding: h[s,d] = wte[ids[s],d] + wpe[s,d]
// ---------------------------------------------------------------------------
__global__ void embed_k(const int* __restrict__ ids, const float* __restrict__ wte,
                        const float* __restrict__ wpe, float* __restrict__ h) {
  int i = blockIdx.x * blockDim.x + threadIdx.x;
  if (i >= S_ * D_) return;
  int s = i / D_, d = i - s * D_;
  h[i] = wte[(size_t)ids[s] * D_ + d] + wpe[i];
}

// ---------------------------------------------------------------------------
// Unpack bit-weights: byte k of row n -> 8 f16 {-1,+1} at [n, k*8..k*8+7]
// ---------------------------------------------------------------------------
__global__ void unpack_k(const unsigned char* __restrict__ p,
                         _Float16* __restrict__ w, int nbytes) {
  int i = blockIdx.x * blockDim.x + threadIdx.x;
  if (i >= nbytes) return;
  const unsigned int byte = p[i];
  v8h o;
#pragma unroll
  for (int j = 0; j < 8; ++j)
    o[j] = ((byte >> j) & 1) ? (_Float16)1.0f : (_Float16)-1.0f;
  *(v8h*)(w + (size_t)i * 8) = o;
}

// ---------------------------------------------------------------------------
// f32 -> f16 conversion (wte), 8 elements per thread
// ---------------------------------------------------------------------------
__global__ void cvt_h_k(const float* __restrict__ src, _Float16* __restrict__ dst,
                        int n8) {
  int i = blockIdx.x * blockDim.x + threadIdx.x;
  if (i >= n8) return;
  const float* s = src + (size_t)i * 8;
  v8h o;
#pragma unroll
  for (int j = 0; j < 8; ++j) o[j] = (_Float16)s[j];
  *(v8h*)(dst + (size_t)i * 8) = o;
}

// ---------------------------------------------------------------------------
// LayerNorm (one block per row), f32 in -> f16 out
// ---------------------------------------------------------------------------
__global__ void ln_k(const float* __restrict__ h, const float* __restrict__ g,
                     const float* __restrict__ b, _Float16* __restrict__ out) {
  const int row = blockIdx.x;
  const int tid = threadIdx.x;  // 256 threads
  const float* x = h + (size_t)row * D_;
  float s1 = 0.f, s2 = 0.f;
  for (int i = tid; i < D_; i += 256) { float v = x[i]; s1 += v; s2 += v * v; }
  __shared__ float r1[256], r2[256];
  r1[tid] = s1; r2[tid] = s2;
  __syncthreads();
  for (int off = 128; off > 0; off >>= 1) {
    if (tid < off) { r1[tid] += r1[tid + off]; r2[tid] += r2[tid + off]; }
    __syncthreads();
  }
  const float mean = r1[0] * (1.f / D_);
  const float var  = r2[0] * (1.f / D_) - mean * mean;
  const float rstd = rsqrtf(var + 1e-5f);
  for (int i = tid; i < D_; i += 256)
    out[(size_t)row * D_ + i] = (_Float16)((x[i] - mean) * rstd * g[i] + b[i]);
}

// ---------------------------------------------------------------------------
// Staged WMMA GEMM: C[M,N] = A[M,K](f16) * B[N,K]^T + bias
// Block = 128 threads (4 waves), tile 128 rows x 64 cols.
// B tile (64n x 32k = 4KB) staged in LDS, shared by all 4 waves; async
// double-buffered (GLOBAL_LOAD_ASYNC_TO_LDS_B128 + s_wait_asynccnt) when
// available, plain load/ds-store otherwise.
//  OUTMODE: 0 store f32, 1 residual-add f32, 2 store f16 ; ACT: 1 = GELU
//  GUARD: bounds-check N (lm_head tail); loads clamp to row N-1.
// ---------------------------------------------------------------------------
template <int OUTMODE, int ACT, int GUARD>
__global__ void __launch_bounds__(128)
gemm_k(const _Float16* __restrict__ A, const _Float16* __restrict__ B,
       const float* __restrict__ bias, float* __restrict__ Cf,
       _Float16* __restrict__ Ch, int M, int N, int K) {
  const int tid  = threadIdx.x;
  const int lane = tid & 31;
  const int wid  = tid >> 5;
  const int ln = lane & 15;
  const int kb = (lane >> 4) * 8;
  const int n0 = blockIdx.x * 64;
  const int m0 = blockIdx.y * 128 + wid * 32;

  __shared__ _Float16 sb[2 * 64 * 32];  // double-buffered B tile (2 x 4KB)
  const int bufHalves = 64 * 32;

  // This thread's two 16B chunks of the B tile: chunk c -> n=c>>2, koff=(c&3)*8
  const int c1 = tid + 128;
  int bn0 = n0 + (tid >> 2);
  int bn1 = n0 + (c1 >> 2);
  if (GUARD) {
    bn0 = bn0 < N - 1 ? bn0 : N - 1;
    bn1 = bn1 < N - 1 ? bn1 : N - 1;
  }
  const _Float16* gb0 = B + (size_t)bn0 * K + (tid & 3) * 8;
  const _Float16* gb1 = B + (size_t)bn1 * K + (c1 & 3) * 8;
  _Float16* lb0 = &sb[0] + tid * 8;  // LDS byte offset = chunk*16
  _Float16* lb1 = &sb[0] + c1 * 8;

  const _Float16* a0 = A + (size_t)(m0 + ln) * K + kb;
  const _Float16* a1 = a0 + (size_t)16 * K;

  v8f acc[2][4];
#pragma unroll
  for (int i = 0; i < 2; ++i)
#pragma unroll
    for (int t = 0; t < 4; ++t)
#pragma unroll
      for (int r = 0; r < 8; ++r) acc[i][t][r] = 0.f;

  auto compute = [&](const _Float16* __restrict__ bb, int k0) {
    v16h a[2];
    {
      v8h lo = *(const v8h*)(a0 + k0);
      v8h hi = *(const v8h*)(a0 + k0 + 16);
#pragma unroll
      for (int j = 0; j < 8; ++j) { a[0][j] = lo[j]; a[0][j + 8] = hi[j]; }
    }
    {
      v8h lo = *(const v8h*)(a1 + k0);
      v8h hi = *(const v8h*)(a1 + k0 + 16);
#pragma unroll
      for (int j = 0; j < 8; ++j) { a[1][j] = lo[j]; a[1][j + 8] = hi[j]; }
    }
#pragma unroll
    for (int t = 0; t < 4; ++t) {
      const _Float16* bp = bb + (16 * t + ln) * 32 + kb;
      v8h lo = *(const v8h*)(bp);
      v8h hi = *(const v8h*)(bp + 16);
      v16h b;
#pragma unroll
      for (int j = 0; j < 8; ++j) { b[j] = lo[j]; b[j + 8] = hi[j]; }
      acc[0][t] = wmma_f16(a[0], b, acc[0][t]);
      acc[1][t] = wmma_f16(a[1], b, acc[1][t]);
    }
  };

  const int nsteps = K >> 5;
#if ASYNC_LDS
  auto issue = [&](int s) {
    const int k0 = s << 5;
    const int bsel = (s & 1) * bufHalves;
    __builtin_amdgcn_global_load_async_to_lds_b128((gv4p)(gb0 + k0),
                                                   (lv4p)(lb0 + bsel), 0, 0);
    __builtin_amdgcn_global_load_async_to_lds_b128((gv4p)(gb1 + k0),
                                                   (lv4p)(lb1 + bsel), 0, 0);
  };
  issue(0);
  for (int s = 0; s < nsteps; ++s) {
    if (s + 1 < nsteps) { issue(s + 1); wait_async_le2(); }
    else                { wait_async_le0(); }
    __syncthreads();  // stage-s B tile fully in LDS (all waves waited)
    compute(&sb[(s & 1) * bufHalves], s << 5);
    __syncthreads();  // all reads of this buffer done before it is reloaded
  }
#else
  for (int s = 0; s < nsteps; ++s) {
    const int k0 = s << 5;
    v8h t0 = *(const v8h*)(gb0 + k0);
    v8h t1 = *(const v8h*)(gb1 + k0);
    __syncthreads();  // previous stage's reads done
    *(v8h*)lb0 = t0;
    *(v8h*)lb1 = t1;
    __syncthreads();
    compute(&sb[0], k0);
  }
#endif

  // Epilogue: acc[i][t][r] -> row m0 + 16i + r + 8*(lane>>4), col n0 + 16t + ln
#pragma unroll
  for (int i = 0; i < 2; ++i) {
    const int rbase = m0 + 16 * i + ((lane >> 4) << 3);
#pragma unroll
    for (int t = 0; t < 4; ++t) {
      const int col = n0 + 16 * t + ln;
      if (GUARD && col >= N) continue;
      const float bv = bias ? bias[col] : 0.f;
#pragma unroll
      for (int r = 0; r < 8; ++r) {
        float v = acc[i][t][r] + bv;
        if (ACT == 1) v = gelu_f(v);
        const size_t idx = (size_t)(rbase + r) * N + col;
        if (OUTMODE == 0)      Cf[idx] = v;
        else if (OUTMODE == 1) Cf[idx] += v;
        else                   Ch[idx] = (_Float16)v;
      }
    }
  }
}

// ---------------------------------------------------------------------------
// Fallback lm_head GEMM (B = wte f32, converted on the fly). One wave / tile.
// ---------------------------------------------------------------------------
__global__ void gemm_f32b_k(const _Float16* __restrict__ A,
                            const float* __restrict__ Bf,
                            float* __restrict__ Cf, int M, int N, int K) {
  const int lane = threadIdx.x;
  const int ln = lane & 15;
  const int kb = (lane >> 4) * 8;
  const int n0 = blockIdx.x * 64;
  const int m0 = blockIdx.y * 32;
  v8f acc[2][4];
#pragma unroll
  for (int i = 0; i < 2; ++i)
#pragma unroll
    for (int t = 0; t < 4; ++t)
#pragma unroll
      for (int r = 0; r < 8; ++r) acc[i][t][r] = 0.f;
  const _Float16* a0 = A + (size_t)(m0 + ln) * K + kb;
  const _Float16* a1 = a0 + (size_t)16 * K;
  for (int k0 = 0; k0 < K; k0 += 32) {
    v16h a[2];
    {
      v8h lo = *(const v8h*)(a0 + k0);
      v8h hi = *(const v8h*)(a0 + k0 + 16);
#pragma unroll
      for (int j = 0; j < 8; ++j) { a[0][j] = lo[j]; a[0][j + 8] = hi[j]; }
    }
    {
      v8h lo = *(const v8h*)(a1 + k0);
      v8h hi = *(const v8h*)(a1 + k0 + 16);
#pragma unroll
      for (int j = 0; j < 8; ++j) { a[1][j] = lo[j]; a[1][j + 8] = hi[j]; }
    }
#pragma unroll
    for (int t = 0; t < 4; ++t) {
      const int n = n0 + 16 * t + ln;
      v16h b;
      if (n < N) {
        const float* bf = Bf + (size_t)n * K + k0 + kb;
#pragma unroll
        for (int j = 0; j < 8; ++j) {
          b[j] = (_Float16)bf[j];
          b[j + 8] = (_Float16)bf[j + 16];
        }
      } else {
#pragma unroll
        for (int j = 0; j < 16; ++j) b[j] = (_Float16)0.0f;
      }
      acc[0][t] = wmma_f16(a[0], b, acc[0][t]);
      acc[1][t] = wmma_f16(a[1], b, acc[1][t]);
    }
  }
#pragma unroll
  for (int i = 0; i < 2; ++i) {
    const int rbase = m0 + 16 * i + ((lane >> 4) << 3);
#pragma unroll
    for (int t = 0; t < 4; ++t) {
      const int col = n0 + 16 * t + ln;
      if (col >= N) continue;
#pragma unroll
      for (int r = 0; r < 8; ++r)
        Cf[(size_t)(rbase + r) * N + col] = acc[i][t][r];
    }
  }
}

// ---------------------------------------------------------------------------
// Flash-style causal attention. One wave per (16-query tile, head).
// Q pre-scaled by 1/sqrt(HD); only the single diagonal chunk is masked.
// ---------------------------------------------------------------------------
__global__ void attn_k(const _Float16* __restrict__ qkv, _Float16* __restrict__ y) {
  const int lane = threadIdx.x;
  const int ln = lane & 15;
  const int kb = (lane >> 4) * 8;
  const int q0 = blockIdx.x * 16;
  const int head = blockIdx.y;
  const size_t QO = (size_t)head * HD_;

  // Q fragments, pre-scaled by 0.125 = 1/sqrt(64)
  const _Float16* qp = qkv + (size_t)(q0 + ln) * (3 * D_) + QO;
  v16h aq[2];
#pragma unroll
  for (int c = 0; c < 2; ++c) {
    v8h lo = *(const v8h*)(qp + 32 * c + kb);
    v8h hi = *(const v8h*)(qp + 32 * c + kb + 16);
#pragma unroll
    for (int j = 0; j < 8; ++j) {
      aq[c][j]     = lo[j] * (_Float16)0.125f;
      aq[c][j + 8] = hi[j] * (_Float16)0.125f;
    }
  }

  v8f acc[4];
#pragma unroll
  for (int t = 0; t < 4; ++t)
#pragma unroll
    for (int r = 0; r < 8; ++r) acc[t][r] = 0.f;
  float mrow[8], lrow[8];
#pragma unroll
  for (int r = 0; r < 8; ++r) { mrow[r] = -3e38f; lrow[r] = 0.f; }

  __shared__ _Float16 pbuf[16 * 32];

  auto do_chunk = [&](int kc, auto mc) {
    constexpr bool MASKED = decltype(mc)::value;
    // ---- scores S[16,32] = Q * K^T (two 16x16 tiles) ----
    v8f sc[2];
#pragma unroll
    for (int t = 0; t < 2; ++t) {
      const _Float16* kp = qkv + (size_t)(kc + 16 * t + ln) * (3 * D_) + D_ + QO;
      v16h bk0, bk1;
      {
        v8h lo = *(const v8h*)(kp + kb);
        v8h hi = *(const v8h*)(kp + kb + 16);
#pragma unroll
        for (int j = 0; j < 8; ++j) { bk0[j] = lo[j]; bk0[j + 8] = hi[j]; }
      }
      {
        v8h lo = *(const v8h*)(kp + 32 + kb);
        v8h hi = *(const v8h*)(kp + 32 + kb + 16);
#pragma unroll
        for (int j = 0; j < 8; ++j) { bk1[j] = lo[j]; bk1[j + 8] = hi[j]; }
      }
      v8f z;
#pragma unroll
      for (int r = 0; r < 8; ++r) z[r] = 0.f;
      z = wmma_f16(aq[0], bk0, z);
      z = wmma_f16(aq[1], bk1, z);
      sc[t] = z;
    }
    // ---- (mask) + online softmax ----
    float pmax[8], psum[8];
#pragma unroll
    for (int r = 0; r < 8; ++r) {
      float s0 = sc[0][r];
      float s1 = sc[1][r];
      if (MASKED) {
        const int qr = q0 + r + ((lane >> 4) << 3);
        if (kc + ln > qr) s0 = -3e38f;
        if (kc + 16 + ln > qr) s1 = -3e38f;
        sc[0][r] = s0; sc[1][r] = s1;
      }
      pmax[r] = fmaxf(s0, s1);
    }
#pragma unroll
    for (int m = 1; m < 16; m <<= 1)
#pragma unroll
      for (int r = 0; r < 8; ++r)
        pmax[r] = fmaxf(pmax[r], __shfl_xor(pmax[r], m, 32));
#pragma unroll
    for (int r = 0; r < 8; ++r) {
      const float mnew = fmaxf(mrow[r], pmax[r]);
      const float cf = __expf(mrow[r] - mnew);
      mrow[r] = mnew;
      const float p0 = __expf(sc[0][r] - mnew);
      const float p1 = __expf(sc[1][r] - mnew);
      sc[0][r] = p0; sc[1][r] = p1;
      psum[r] = p0 + p1;
      lrow[r] *= cf;
#pragma unroll
      for (int t = 0; t < 4; ++t) acc[t][r] *= cf;
    }
#pragma unroll
    for (int m = 1; m < 16; m <<= 1)
#pragma unroll
      for (int r = 0; r < 8; ++r) psum[r] += __shfl_xor(psum[r], m, 32);
#pragma unroll
    for (int r = 0; r < 8; ++r) lrow[r] += psum[r];

    // ---- re-layout P (C-layout -> A-layout) through LDS ----
    __syncthreads();
#pragma unroll
    for (int t = 0; t < 2; ++t)
#pragma unroll
      for (int r = 0; r < 8; ++r)
        pbuf[(r + ((lane >> 4) << 3)) * 32 + 16 * t + ln] = (_Float16)sc[t][r];
    __syncthreads();
    v16h pa;
#pragma unroll
    for (int j = 0; j < 16; ++j) pa[j] = pbuf[ln * 32 + kb + j + (j & 8)];

    // ---- acc += P * V ----
#pragma unroll
    for (int t = 0; t < 4; ++t) {
      const int d = 16 * t + ln;
      v16h bv;
#pragma unroll
      for (int j = 0; j < 16; ++j) {
        const int key = kc + kb + j + (j & 8);
        bv[j] = qkv[(size_t)key * (3 * D_) + 2 * D_ + QO + d];
      }
      acc[t] = wmma_f16(pa, bv, acc[t]);
    }
  };

  const int nfull = q0 >> 5;  // chunks fully below the diagonal
  for (int ch = 0; ch < nfull; ++ch) do_chunk(ch * 32, CFalse{});
  do_chunk(nfull * 32, CTrue{});  // the single diagonal chunk

  // ---- normalize + store ----
#pragma unroll
  for (int t = 0; t < 4; ++t)
#pragma unroll
    for (int r = 0; r < 8; ++r) {
      const int row = q0 + r + ((lane >> 4) << 3);
      y[(size_t)row * D_ + QO + 16 * t + ln] = (_Float16)(acc[t][r] / lrow[r]);
    }
}

// ---------------------------------------------------------------------------
extern "C" void kernel_launch(void* const* d_in, const int* in_sizes, int n_in,
                              void* d_out, int out_size, void* d_ws, size_t ws_size,
                              hipStream_t stream) {
  (void)in_sizes; (void)n_in; (void)out_size;
  const int*           ids  = (const int*)d_in[0];
  const float*         wte  = (const float*)d_in[1];
  const float*         wpe  = (const float*)d_in[2];
  const float*         ln1g = (const float*)d_in[3];
  const float*         ln1b = (const float*)d_in[4];
  const float*         ln2g = (const float*)d_in[5];
  const float*         ln2b = (const float*)d_in[6];
  const unsigned char* qkvp = (const unsigned char*)d_in[7];
  const float*         qkvb = (const float*)d_in[8];
  const unsigned char* app  = (const unsigned char*)d_in[9];
  const float*         apb  = (const float*)d_in[10];
  const unsigned char* fcp  = (const unsigned char*)d_in[11];
  const float*         fcb  = (const float*)d_in[12];
  const unsigned char* mpp  = (const unsigned char*)d_in[13];
  const float*         mpb  = (const float*)d_in[14];
  const float*         lnfg = (const float*)d_in[15];
  const float*         lnfb = (const float*)d_in[16];

  char* w = (char*)d_ws;
  auto alloc = [&](size_t bytes) -> char* {
    char* p = w;
    w += (bytes + 255) & ~(size_t)255;
    return p;
  };
  float*    h     = (float*)   alloc(sizeof(float)    * S_ * D_);
  _Float16* x_h   = (_Float16*)alloc(sizeof(_Float16) * S_ * D_);
  _Float16* qkv_h = (_Float16*)alloc(sizeof(_Float16) * S_ * 3 * D_);
  _Float16* y_h   = (_Float16*)alloc(sizeof(_Float16) * S_ * D_);
  _Float16* fc_h  = (_Float16*)alloc(sizeof(_Float16) * S_ * F_);
  _Float16* w_qkv = (_Float16*)alloc(sizeof(_Float16) * 3 * D_ * D_);
  _Float16* w_pr  = (_Float16*)alloc(sizeof(_Float16) * D_ * D_);
  _Float16* w_fc  = (_Float16*)alloc(sizeof(_Float16) * F_ * D_);
  _Float16* w_mp  = (_Float16*)alloc(sizeof(_Float16) * D_ * F_);
  const size_t used = (size_t)(w - (char*)d_ws);
  const size_t wte_bytes =
      ((sizeof(_Float16) * (size_t)V_ * D_) + 255) & ~(size_t)255;
  const bool fast_lm = (used + wte_bytes) <= ws_size;
  _Float16* wte_h = fast_lm ? (_Float16*)alloc(sizeof(_Float16) * (size_t)V_ * D_)
                            : nullptr;

  embed_k<<<(S_ * D_ + 255) / 256, 256, 0, stream>>>(ids, wte, wpe, h);
  if (fast_lm) {
    const int n8 = (V_ * D_) / 8;
    cvt_h_k<<<(n8 + 255) / 256, 256, 0, stream>>>(wte, wte_h, n8);
  }

  const int QKV_B = 3 * D_ * (D_ / 8);
  const int PR_B  = D_ * (D_ / 8);
  const int FC_B  = F_ * (D_ / 8);
  const int MP_B  = D_ * (F_ / 8);

  for (int l = 0; l < L_; ++l) {
    unpack_k<<<(QKV_B + 255) / 256, 256, 0, stream>>>(qkvp + (size_t)l * QKV_B, w_qkv, QKV_B);
    unpack_k<<<(PR_B  + 255) / 256, 256, 0, stream>>>(app  + (size_t)l * PR_B,  w_pr,  PR_B);
    unpack_k<<<(FC_B  + 255) / 256, 256, 0, stream>>>(fcp  + (size_t)l * FC_B,  w_fc,  FC_B);
    unpack_k<<<(MP_B  + 255) / 256, 256, 0, stream>>>(mpp  + (size_t)l * MP_B,  w_mp,  MP_B);

    ln_k<<<S_, 256, 0, stream>>>(h, ln1g + l * D_, ln1b + l * D_, x_h);
    gemm_k<2, 0, 0><<<dim3(3 * D_ / 64, S_ / 128), 128, 0, stream>>>(
        x_h, w_qkv, qkvb + (size_t)l * 3 * D_, nullptr, qkv_h, S_, 3 * D_, D_);
    attn_k<<<dim3(S_ / 16, H_), 32, 0, stream>>>(qkv_h, y_h);
    gemm_k<1, 0, 0><<<dim3(D_ / 64, S_ / 128), 128, 0, stream>>>(
        y_h, w_pr, apb + (size_t)l * D_, h, nullptr, S_, D_, D_);
    ln_k<<<S_, 256, 0, stream>>>(h, ln2g + l * D_, ln2b + l * D_, x_h);
    gemm_k<2, 1, 0><<<dim3(F_ / 64, S_ / 128), 128, 0, stream>>>(
        x_h, w_fc, fcb + (size_t)l * F_, nullptr, fc_h, S_, F_, D_);
    gemm_k<1, 0, 0><<<dim3(D_ / 64, S_ / 128), 128, 0, stream>>>(
        fc_h, w_mp, mpb + (size_t)l * D_, h, nullptr, S_, D_, F_);
  }

  ln_k<<<S_, 256, 0, stream>>>(h, lnfg, lnfb, x_h);
  if (fast_lm) {
    gemm_k<0, 0, 1><<<dim3((V_ + 63) / 64, S_ / 128), 128, 0, stream>>>(
        x_h, wte_h, nullptr, (float*)d_out, nullptr, S_, V_, D_);
  } else {
    gemm_f32b_k<<<dim3((V_ + 63) / 64, S_ / 32), 32, 0, stream>>>(
        x_h, wte, (float*)d_out, S_, V_, D_);
  }
}
